// MMoELayer_42253888258347
// MI455X (gfx1250) — compile-verified, matrix operand order checked by reading, not run on hardware
//
#include <hip/hip_runtime.h>
#include <stdint.h>

// MMoE fused: B=16384, D=1024, H0=1024, H1=512, E=8, T=2
#define B_   16384
#define D_   1024
#define H0_  1024
#define H1_  512
#define E_   8
#define T_   2
#define BM   32              // batch rows per block in fused kernel
#define NBLK (B_ / BM)       // 512 blocks

typedef __attribute__((ext_vector_type(16))) __bf16 v16bf;
typedef __attribute__((ext_vector_type(8)))  float  v8f;

union Frag {
  v16bf v;
  uint4 q[2];
  unsigned short s[16];
};

__device__ __forceinline__ unsigned short f2bf(float f) {
  union { float f; unsigned int u; } x; x.f = f;
  unsigned int r = x.u + 0x7FFFu + ((x.u >> 16) & 1u);  // round-to-nearest-even
  return (unsigned short)(r >> 16);
}

// K index inside a 16x32 bf16 A-fragment (ISA 7.12.2): element i, lane-half h
__device__ __forceinline__ int kmap(int i, int half) {
  return ((i >> 3) << 4) + (half << 3) + (i & 7);
}

// ---------------- prep: x [B,D] f32 -> bf16 A-fragment-linear ----------------
// layout: ((bt*(D/32)+kt)*32 + lane)*16 + i  ;  elem = x[bt*16+(lane&15)][kt*32+kmap(i,lane>>4)]
__global__ __launch_bounds__(256) void k_conv_x(const float* __restrict__ x,
                                                unsigned short* __restrict__ xF) {
  int id = blockIdx.x * 256 + threadIdx.x;        // B/16 * D/32 * 32 = 1,048,576
  int lane = id & 31;
  int kt   = (id >> 5) & (D_ / 32 - 1);
  int bt   = id >> 10;
  int row  = bt * 16 + (lane & 15);
  int half = lane >> 4;
  unsigned short tmp[16];
#pragma unroll
  for (int i = 0; i < 16; ++i) {
    int k = kt * 32 + kmap(i, half);
    tmp[i] = f2bf(x[(size_t)row * D_ + k]);
  }
  uint4* dst = (uint4*)(xF + (size_t)id * 16);
  dst[0] = *(const uint4*)&tmp[0];
  dst[1] = *(const uint4*)&tmp[8];
}

// -------- prep: W0 [E,D,H0] f32 -> bf16 B-fragment-linear --------
// layout: ((((e*(D/32)+kt)*(H0/16)+nt)*32 + lane)*16 + i)
// elem = W0[e][kt*32+kmap(i,lane>>4)][nt*16+(lane&15)]
__global__ __launch_bounds__(256) void k_conv_w0(const float* __restrict__ W0,
                                                 unsigned short* __restrict__ w0F) {
  int id = blockIdx.x * 256 + threadIdx.x;        // E*32*64*32 = 524,288
  int lane = id & 31;
  int nt   = (id >> 5) & (H0_ / 16 - 1);
  int kt   = (id >> 11) & (D_ / 32 - 1);
  int e    = id >> 16;
  int n    = nt * 16 + (lane & 15);
  int half = lane >> 4;
  unsigned short tmp[16];
#pragma unroll
  for (int i = 0; i < 16; ++i) {
    int k = kt * 32 + kmap(i, half);
    tmp[i] = f2bf(W0[((size_t)e * D_ + k) * H0_ + n]);
  }
  uint4* dst = (uint4*)(w0F + (size_t)id * 16);
  dst[0] = *(const uint4*)&tmp[0];
  dst[1] = *(const uint4*)&tmp[8];
}

// -------- prep: W1 [E,H0,H1] f32 -> bf16 B-fragment-linear --------
__global__ __launch_bounds__(256) void k_conv_w1(const float* __restrict__ W1,
                                                 unsigned short* __restrict__ w1F) {
  int id = blockIdx.x * 256 + threadIdx.x;        // E*32*32*32 = 262,144
  int lane = id & 31;
  int nt   = (id >> 5) & (H1_ / 16 - 1);
  int kt   = (id >> 10) & (H0_ / 32 - 1);
  int e    = id >> 15;
  int n    = nt * 16 + (lane & 15);
  int half = lane >> 4;
  unsigned short tmp[16];
#pragma unroll
  for (int i = 0; i < 16; ++i) {
    int k = kt * 32 + kmap(i, half);
    tmp[i] = f2bf(W1[((size_t)e * H0_ + k) * H1_ + n]);
  }
  uint4* dst = (uint4*)(w1F + (size_t)id * 16);
  dst[0] = *(const uint4*)&tmp[0];
  dst[1] = *(const uint4*)&tmp[8];
}

// ---------------- gates: softmax_e( x[b]·Wg[t,:,e] + bg[t,e] ) ----------------
__global__ __launch_bounds__(256) void k_gates(const float* __restrict__ x,
                                               const float* __restrict__ Wg,
                                               const float* __restrict__ bg,
                                               float* __restrict__ g) {
  int id = blockIdx.x * 256 + threadIdx.x;        // T*B = 32768
  int b = id & (B_ - 1);
  int t = id >> 14;
  float acc[E_];
#pragma unroll
  for (int e = 0; e < E_; ++e) acc[e] = bg[t * E_ + e];
  const float* xr = x + (size_t)b * D_;
  const float* wr = Wg + (size_t)t * D_ * E_;
  for (int d = 0; d < D_; ++d) {
    float xv = xr[d];
#pragma unroll
    for (int e = 0; e < E_; ++e) acc[e] = fmaf(xv, wr[(size_t)d * E_ + e], acc[e]);
  }
  float m = acc[0];
#pragma unroll
  for (int e = 1; e < E_; ++e) m = fmaxf(m, acc[e]);
  float s = 0.f;
#pragma unroll
  for (int e = 0; e < E_; ++e) { acc[e] = __expf(acc[e] - m); s += acc[e]; }
  float inv = 1.f / s;
#pragma unroll
  for (int e = 0; e < E_; ++e) g[((size_t)t * B_ + b) * E_ + e] = acc[e] * inv;
}

// ---------------- fused experts + gated combine ----------------
// Block = 256 threads (8 waves), BM=32 batch rows. All intermediates live in LDS.
__global__ __launch_bounds__(256) void k_mmoe(const unsigned short* __restrict__ xF,
                                              const unsigned short* __restrict__ w0F,
                                              const unsigned short* __restrict__ w1F,
                                              const float* __restrict__ b0,
                                              const float* __restrict__ b1,
                                              const float* __restrict__ gates,
                                              float* __restrict__ out) {
  __shared__ unsigned short xs[BM / 16][D_ / 32][32][16];   // 64 KB  (A-frags of x)
  __shared__ unsigned short hs[BM / 16][H0_ / 32][32][16];  // 64 KB  (A-frags of h)
  __shared__ float outs[T_][BM][H1_];                       // 128 KB (gated accum)
  __shared__ float gs[T_][BM][E_];                          // 2 KB

  const int tid  = threadIdx.x;
  const int lane = tid & 31;
  const int wave = tid >> 5;
  const int blk  = blockIdx.x;
  const int row0 = blk * BM;
  const int half = lane >> 4;
  const int ncol = lane & 15;

  // zero output accumulators
  float* outsF = &outs[0][0][0];
  for (int i = tid; i < T_ * BM * H1_; i += 256) outsF[i] = 0.f;
  // stage gates for this row tile
  for (int i = tid; i < T_ * BM * E_; i += 256) {
    int t = i / (BM * E_);
    int r = (i / E_) % BM;
    int e = i % E_;
    gs[t][r][e] = gates[((size_t)t * B_ + row0 + r) * E_ + e];
  }
  // stage x fragments: contiguous 64 KB block copy
  {
    const uint4* src = (const uint4*)(xF + (size_t)blk * BM * D_);
    uint4* dst = (uint4*)&xs[0][0][0][0];
    for (int i = tid; i < BM * D_ / 8; i += 256) dst[i] = src[i];
  }
  __syncthreads();

  const v8f vzero = {0.f, 0.f, 0.f, 0.f, 0.f, 0.f, 0.f, 0.f};

  for (int e = 0; e < E_; ++e) {
    // ---- GEMM1: h[BM,1024] = relu(x @ W0[e] + b0[e]); wave owns 128 cols ----
    for (int rt = 0; rt < BM / 16; ++rt) {
      v8f acc[8];
#pragma unroll
      for (int j = 0; j < 8; ++j) acc[j] = vzero;
      for (int kt = 0; kt < D_ / 32; ++kt) {
        Frag a;
        const uint4* ap = (const uint4*)&xs[rt][kt][lane][0];
        a.q[0] = ap[0]; a.q[1] = ap[1];
        const unsigned short* wp =
            w0F + ((((size_t)e * (D_ / 32) + kt) * (H0_ / 16) + wave * 8) * 32 + lane) * 16;
#pragma unroll
        for (int j = 0; j < 8; ++j) {
          Frag bfr;
          const uint4* bp = (const uint4*)(wp + (size_t)j * 512);
          bfr.q[0] = bp[0]; bfr.q[1] = bp[1];
          acc[j] = __builtin_amdgcn_wmma_f32_16x16x32_bf16(
              false, a.v, false, bfr.v, (short)0, acc[j], false, false);
        }
      }
      // epilogue: bias + relu, repack tile into hs as GEMM2 A-fragments
#pragma unroll
      for (int j = 0; j < 8; ++j) {
        int n = (wave * 8 + j) * 16 + ncol;   // H0 column (fixed per lane)
        float bias = b0[e * H0_ + n];
        int kt2 = n >> 5;
        int kin = n & 31;
        int g16 = (kin >> 4) & 1;
        int rem = kin & 15;
        int h2  = rem >> 3;
        int ii  = g16 * 8 + (rem & 7);
#pragma unroll
        for (int v = 0; v < 8; ++v) {
          int m = v + half * 8;               // row inside 16-row tile
          float val = fmaxf(acc[j][v] + bias, 0.f);
          hs[rt][kt2][h2 * 16 + m][ii] = f2bf(val);
        }
      }
    }
    __syncthreads();

    // ---- GEMM2 + gated accumulate: eo = relu(h @ W1[e] + b1[e]); wave owns 64 cols ----
    for (int rt = 0; rt < BM / 16; ++rt) {
      v8f acc2[4];
#pragma unroll
      for (int j = 0; j < 4; ++j) acc2[j] = vzero;
      for (int kt = 0; kt < H0_ / 32; ++kt) {
        Frag a;
        const uint4* ap = (const uint4*)&hs[rt][kt][lane][0];
        a.q[0] = ap[0]; a.q[1] = ap[1];
        const unsigned short* wp =
            w1F + ((((size_t)e * (H0_ / 32) + kt) * (H1_ / 16) + wave * 4) * 32 + lane) * 16;
#pragma unroll
        for (int j = 0; j < 4; ++j) {
          Frag bfr;
          const uint4* bp = (const uint4*)(wp + (size_t)j * 512);
          bfr.q[0] = bp[0]; bfr.q[1] = bp[1];
          acc2[j] = __builtin_amdgcn_wmma_f32_16x16x32_bf16(
              false, a.v, false, bfr.v, (short)0, acc2[j], false, false);
        }
      }
#pragma unroll
      for (int j = 0; j < 4; ++j) {
        int n = (wave * 4 + j) * 16 + ncol;   // H1 column (disjoint per wave)
        float bias = b1[e * H1_ + n];
#pragma unroll
        for (int v = 0; v < 8; ++v) {
          int r = rt * 16 + v + half * 8;
          float val = fmaxf(acc2[j][v] + bias, 0.f);
          outs[0][r][n] += gs[0][r][e] * val;
          outs[1][r][n] += gs[1][r][e] * val;
        }
      }
    }
    __syncthreads();
  }

  // write out: tuple order -> out[t*B*H1 + b*H1 + n]
  for (int i = tid; i < T_ * BM * H1_; i += 256) {
    int t = i / (BM * H1_);
    int r = (i / H1_) % BM;
    int n = i % H1_;
    out[((size_t)t * B_ + row0 + r) * H1_ + n] = outs[t][r][n];
  }
}

extern "C" void kernel_launch(void* const* d_in, const int* in_sizes, int n_in,
                              void* d_out, int out_size, void* d_ws, size_t ws_size,
                              hipStream_t stream) {
  const float* x  = (const float*)d_in[0];   // [B,D]
  const float* W0 = (const float*)d_in[1];   // [E,D,H0]
  const float* b0 = (const float*)d_in[2];   // [E,H0]
  const float* W1 = (const float*)d_in[3];   // [E,H0,H1]
  const float* b1 = (const float*)d_in[4];   // [E,H1]
  const float* Wg = (const float*)d_in[5];   // [T,D,E]
  const float* bg = (const float*)d_in[6];   // [T,E]
  float* out = (float*)d_out;                // T*[B,H1] concatenated

  char* ws = (char*)d_ws;
  size_t off = 0;
  unsigned short* xF  = (unsigned short*)(ws + off); off += (size_t)B_ * D_ * 2;        // 33.5 MB
  unsigned short* w0F = (unsigned short*)(ws + off); off += (size_t)E_ * D_ * H0_ * 2;  // 16.8 MB
  unsigned short* w1F = (unsigned short*)(ws + off); off += (size_t)E_ * H0_ * H1_ * 2; // 8.4 MB
  float* gbuf = (float*)(ws + off);                                                     // 1 MB

  k_conv_x <<<dim3((B_ / 16) * (D_ / 32) * 32 / 256), dim3(256), 0, stream>>>(x, xF);
  k_conv_w0<<<dim3(E_ * (D_ / 32) * (H0_ / 16) * 32 / 256), dim3(256), 0, stream>>>(W0, w0F);
  k_conv_w1<<<dim3(E_ * (H0_ / 32) * (H1_ / 16) * 32 / 256), dim3(256), 0, stream>>>(W1, w1F);
  k_gates  <<<dim3(T_ * B_ / 256), dim3(256), 0, stream>>>(x, Wg, bg, gbuf);
  k_mmoe   <<<dim3(NBLK), dim3(256), 0, stream>>>(xF, w0F, w1F, b0, b1, gbuf, out);
}
